// MapModule_83159156785488
// MI455X (gfx1250) — matrix operand order repaired and checked by reading, not compiled, not run on hardware
//
#include <hip/hip_runtime.h>
#include <hip/hip_bf16.h>

typedef __attribute__((ext_vector_type(16))) _Float16 v16h;
typedef __attribute__((ext_vector_type(8)))  float    v8f;

#define HW3 (512*1024)
#define HW2 (256*512)
#define HW1 (128*256)
#define HW0 (64*128)

// xor-16 lane exchange via LDS swizzle (group-of-32, xor_mask=0x10, and=0x1f)
__device__ __forceinline__ float xor16(float v) {
  return __int_as_float(__builtin_amdgcn_ds_swizzle(__float_as_int(v), 0x401F));
}

__device__ __forceinline__ float sigmoidf(float x) {
  return 1.0f / (1.0f + __expf(-x));
}

// Kernel 1: materialize combined atlas, PIXEL-major: ws[pix*64 + c].
// thread t -> (pix = t>>4, c4 = t&15), writes one float4; lanes 0..15 of a
// 16-thread group cover one pixel's 64 channels -> fully coalesced stores.
__global__ void __launch_bounds__(256) build_atlas_kernel(
    const float* __restrict__ a0, const float* __restrict__ a1,
    const float* __restrict__ a2, const float* __restrict__ a3,
    float* __restrict__ ws) {
  int t   = blockIdx.x * blockDim.x + threadIdx.x;
  int c4  = t & 15;
  int pix = t >> 4;
  if (pix >= HW3) return;
  int y = pix >> 10, x = pix & 1023;
  int o2 = (y >> 1) * 512 + (x >> 1);
  int o1 = (y >> 2) * 256 + (x >> 2);
  int o0 = (y >> 3) * 128 + (x >> 3);
  int c = c4 << 2;
  float4 v;
  float* vp = (float*)&v;
#pragma unroll
  for (int j = 0; j < 4; ++j) {
    int ch = c + j;
    vp[j] = 0.25f * (a3[ch * HW3 + pix] + a2[ch * HW2 + o2] +
                     a1[ch * HW1 + o1] + a0[ch * HW0 + o0]);
  }
  ((float4*)ws)[(size_t)pix * 16 + c4] = v;
}

// One branch of the field: LN1 -> WMMA(32x16) -> leaky -> LN2 -> WMMA(16xO) -> leaky.
// raw[16]: this lane's 16 of the sample's 32 branch features
//          (lane<16: ch 0..15, lane>=16: ch 16..31, same k order as aw1).
__device__ __forceinline__ v8f run_branch(
    const float (&raw)[16], int half,
    const float* __restrict__ g1, const float* __restrict__ b1v,
    const float* __restrict__ g2, const float* __restrict__ b2v,
    v16h aw1, v8f cb1, v16h aw2, v8f cb2) {
  // ---- LayerNorm over 32 features (16 here + 16 in partner lane) ----
  float s = 0.f, q = 0.f;
#pragma unroll
  for (int i = 0; i < 16; ++i) { s += raw[i]; q += raw[i] * raw[i]; }
  s += xor16(s);  q += xor16(q);
  float mu  = s * (1.f / 32.f);
  float var = q * (1.f / 32.f) - mu * mu;
  float rs  = rsqrtf(var + 1e-5f);
  int cb = half * 16;
  v16h b;
#pragma unroll
  for (int i = 0; i < 16; ++i) {
    float v = (raw[i] - mu) * rs * g1[cb + i] + b1v[cb + i];
    b[i] = (_Float16)v;
  }
  // H' = W1^T (16x32) * X^T (32x16) + b1 : lane holds hidden half*8+r of sample lane%16
  v8f h = __builtin_amdgcn_wmma_f32_16x16x32_f16(
      false, aw1, false, b, (short)0, cb1, false, false);
  // leaky relu + LN2 stats (16 hidden: 8 here + 8 in partner lane)
  float hv[8];
  float s2 = 0.f, q2 = 0.f;
#pragma unroll
  for (int r = 0; r < 8; ++r) {
    float v = h[r];
    v = v > 0.f ? v : 0.01f * v;
    hv[r] = v;  s2 += v;  q2 += v * v;
  }
  s2 += xor16(s2);  q2 += xor16(q2);
  float mu2  = s2 * (1.f / 16.f);
  float var2 = q2 * (1.f / 16.f) - mu2 * mu2;
  float rs2  = rsqrtf(var2 + 1e-5f);
  int hb = half * 8;
  // Build layer-2 B (16x16 padded to K=32; A columns K>=16 are zero, so the
  // B slots 8..15 of every lane are don't-care).
  v16h b2;
#pragma unroll
  for (int r = 0; r < 8; ++r) {
    float own = (hv[r] - mu2) * rs2 * g2[hb + r] + b2v[hb + r];
    float oth = xor16(own);                // partner's hidden slice
    b2[r]     = (_Float16)own;             // lane<16: K=r      (hidden r)
    b2[8 + r] = (_Float16)oth;             // lane<16: K=16+r (A=0)    (hidden 8+r)
  }
  v8f o = __builtin_amdgcn_wmma_f32_16x16x32_f16(
      false, aw2, false, b2, (short)0, cb2, false, false);
#pragma unroll
  for (int r = 0; r < 8; ++r) {
    float v = o[r];
    o[r] = v > 0.f ? v : 0.01f * v;
  }
  return o;
}

#define PUT4(arr, off, v) { arr[(off)] = (v).x; arr[(off)+1] = (v).y; \
                            arr[(off)+2] = (v).z; arr[(off)+3] = (v).w; }

__global__ void __launch_bounds__(256) fused_field_kernel(
    const int*   __restrict__ idx,
    const float* __restrict__ a0,      const float* __restrict__ a1,
    const float* __restrict__ a2,      const float* __restrict__ a3,
    const float* __restrict__ c_ln1_g, const float* __restrict__ c_ln1_b,
    const float* __restrict__ c_w1,    const float* __restrict__ c_b1,
    const float* __restrict__ c_ln2_g, const float* __restrict__ c_ln2_b,
    const float* __restrict__ c_w2,    const float* __restrict__ c_b2,
    const float* __restrict__ d_ln1_g, const float* __restrict__ d_ln1_b,
    const float* __restrict__ d_w1,    const float* __restrict__ d_b1,
    const float* __restrict__ d_ln2_g, const float* __restrict__ d_ln2_b,
    const float* __restrict__ d_w2,    const float* __restrict__ d_b2,
    const float* __restrict__ feat,    // materialized pixel-major atlas, or null
    float* __restrict__ out, int n) {
  constexpr int TPW = 4;                       // tiles (of 16 samples) per wave
  int tid  = blockIdx.x * blockDim.x + threadIdx.x;
  int lane = threadIdx.x & 31;
  int wave = tid >> 5;
  int m    = lane & 15;                        // row / sample-in-tile / N index
  int half = lane >> 4;

  // ---- hoisted A matrices (W^T) and C biases, ISA per-lane layouts ----
  // A slot s   <-> K = half*16 + s (W1) / half*8 + s (W2) ; slot 8+s <-> K = half*16+8+s (W1) / zero (W2)
  v16h a1d, a1c, a2d, a2c;
#pragma unroll
  for (int s = 0; s < 8; ++s) {
    int k0 = half * 8 + s;
    int k1 = 16 + half * 8 + s;
    a1d[s]     = (_Float16)d_w1[(half * 16 + s) * 16 + m];
    a1d[8 + s] = (_Float16)d_w1[(half * 16 + 8 + s) * 16 + m];
    a1c[s]     = (_Float16)c_w1[(half * 16 + s) * 16 + m];
    a1c[8 + s] = (_Float16)c_w1[(half * 16 + 8 + s) * 16 + m];
    a2d[s]     = (_Float16)((m < 4) ? d_w2[k0 * 4 + m] : 0.f);
    a2d[8 + s] = (_Float16)0.f;                // K>=16 zero-padded
    a2c[s]     = (_Float16)((m < 3) ? c_w2[k0 * 3 + m] : 0.f);
    a2c[8 + s] = (_Float16)0.f;
  }
  v8f c1d, c1c, c2d, c2c;
#pragma unroll
  for (int r = 0; r < 8; ++r) {
    int h = half * 8 + r;                      // hidden / out-unit row index
    c1d[r] = d_b1[h];
    c1c[r] = c_b1[h];
    c2d[r] = (h < 4) ? d_b2[h] : 0.f;
    c2c[r] = (h < 3) ? c_b2[h] : 0.f;
  }

#pragma unroll
  for (int t = 0; t < TPW; ++t) {
    int s0 = (wave * TPW + t) * 16;
    if (s0 >= n) break;                        // uniform per wave
    int sample = s0 + m;
    int sclamp = min(sample, n - 1);
    int pix = idx[sclamp];

    // raw features: branch d = ch half*16..+15 ; branch c = 32 + half*16..+15
    float rawd[16], rawc[16];
    if (feat) {
      const float4* f4 = (const float4*)(feat + (size_t)pix * 64);
      float4 v0 = f4[half * 4 + 0], v1 = f4[half * 4 + 1];
      float4 v2 = f4[half * 4 + 2], v3 = f4[half * 4 + 3];
      PUT4(rawd, 0, v0); PUT4(rawd, 4, v1); PUT4(rawd, 8, v2); PUT4(rawd, 12, v3);
      float4 w0 = f4[8 + half * 4 + 0], w1v = f4[8 + half * 4 + 1];
      float4 w2v = f4[8 + half * 4 + 2], w3 = f4[8 + half * 4 + 3];
      PUT4(rawc, 0, w0); PUT4(rawc, 4, w1v); PUT4(rawc, 8, w2v); PUT4(rawc, 12, w3);
    } else {
      int y = pix >> 10, x = pix & 1023;
      int o2 = (y >> 1) * 512 + (x >> 1);
      int o1 = (y >> 2) * 256 + (x >> 2);
      int o0 = (y >> 3) * 128 + (x >> 3);
#pragma unroll
      for (int i = 0; i < 16; ++i) {
        int chd = half * 16 + i;
        rawd[i] = 0.25f * (a3[chd * HW3 + pix] + a2[chd * HW2 + o2] +
                           a1[chd * HW1 + o1] + a0[chd * HW0 + o0]);
        int chc = 32 + chd;
        rawc[i] = 0.25f * (a3[chc * HW3 + pix] + a2[chc * HW2 + o2] +
                           a1[chc * HW1 + o1] + a0[chc * HW0 + o0]);
      }
    }

    v8f od = run_branch(rawd, half, d_ln1_g, d_ln1_b, d_ln2_g, d_ln2_b,
                        a1d, c1d, a2d, c2d);
    v8f oc = run_branch(rawc, half, c_ln1_g, c_ln1_b, c_ln2_g, c_ln2_b,
                        a1c, c1c, a2c, c2c);

    // lanes 0..15 hold out units 0..7 of their sample; skips (ch0 / ch32..34)
    // are rawd[0] / rawc[0..2] on those same lanes.
    if (half == 0 && sample < n) {
      float dens = sigmoidf(rawd[0] + od[0]);
      float cr   = sigmoidf(rawc[0] + oc[0]);
      float cg   = sigmoidf(rawc[1] + oc[1]);
      float cbv  = sigmoidf(rawc[2] + oc[2]);
      ((float4*)out)[sample] = make_float4(cr, cg, cbv, dens);
    }
  }
}

extern "C" void kernel_launch(void* const* d_in, const int* in_sizes, int n_in,
                              void* d_out, int out_size, void* d_ws, size_t ws_size,
                              hipStream_t stream) {
  const int*   idx     = (const int*)d_in[0];
  const float* a0      = (const float*)d_in[1];
  const float* a1      = (const float*)d_in[2];
  const float* a2      = (const float*)d_in[3];
  const float* a3      = (const float*)d_in[4];
  const float* c_ln1_g = (const float*)d_in[5];
  const float* c_ln1_b = (const float*)d_in[6];
  const float* c_w1    = (const float*)d_in[7];
  const float* c_b1    = (const float*)d_in[8];
  const float* c_ln2_g = (const float*)d_in[9];
  const float* c_ln2_b = (const float*)d_in[10];
  const float* c_w2    = (const float*)d_in[11];
  const float* c_b2    = (const float*)d_in[12];
  const float* d_ln1_g = (const float*)d_in[13];
  const float* d_ln1_b = (const float*)d_in[14];
  const float* d_w1    = (const float*)d_in[15];
  const float* d_b1    = (const float*)d_in[16];
  const float* d_ln2_g = (const float*)d_in[17];
  const float* d_ln2_b = (const float*)d_in[18];
  const float* d_w2    = (const float*)d_in[19];
  const float* d_b2    = (const float*)d_in[20];
  float* out = (float*)d_out;
  int n = in_sizes[0];

  const size_t need = (size_t)HW3 * 64 * sizeof(float);   // 128 MB
  const float* feat = nullptr;
  if (ws_size >= need) {
    int total  = HW3 * 16;                                // one float4 per thread
    int blocks = (total + 255) / 256;
    build_atlas_kernel<<<blocks, 256, 0, stream>>>(a0, a1, a2, a3, (float*)d_ws);
    feat = (const float*)d_ws;
  }

  const int TPW = 4;
  int tiles   = (n + 15) / 16;
  int waves   = (tiles + TPW - 1) / TPW;
  int threads = waves * 32;
  int blocks  = (threads + 255) / 256;
  fused_field_kernel<<<blocks, 256, 0, stream>>>(
      idx, a0, a1, a2, a3,
      c_ln1_g, c_ln1_b, c_w1, c_b1, c_ln2_g, c_ln2_b, c_w2, c_b2,
      d_ln1_g, d_ln1_b, d_w1, d_b1, d_ln2_g, d_ln2_b, d_w2, d_b2,
      feat, out, n);
}
